// CoarsePointMatching_60739427500773
// MI455X (gfx1250) — matrix operand order repaired and checked by reading, not compiled, not run on hardware
//
#include <hip/hip_runtime.h>

// ---------------- constants ----------------
constexpr int CB   = 2;     // batch
constexpr int CN   = 196;   // points
constexpr int CNP  = 197;   // points + bg token
constexpr int CD   = 256;   // model dim
constexpr int CNH  = 4;     // heads
constexpr int CC   = 64;    // head dim
constexpr int CF   = 512;   // ffn dim
constexpr int CNM  = 512;   // model points
constexpr int KPAD = 224;   // 197 padded to multiple of 32 for A*V WMMA
constexpr int NP1  = 6000;
constexpr int NP2  = 300;
constexpr float CTAU  = 0.1f;
constexpr float CTEMP = 0.1f;

#define DEV __device__ __forceinline__

// ---------------- WMMA types ----------------
typedef __bf16 bf16_t;
typedef __attribute__((ext_vector_type(16))) __bf16 v16bf;
typedef __attribute__((ext_vector_type(8)))  float  v8f;
typedef __attribute__((ext_vector_type(4)))  unsigned int u32x4;

union FragAB { v16bf bf; u32x4 q[2]; };
static_assert(sizeof(FragAB) == 32, "frag size");

DEV unsigned short f2bf(float f) {
  unsigned int x = __float_as_uint(f);
  unsigned int r = x + 0x7FFFu + ((x >> 16) & 1u);   // RNE
  return (unsigned short)(r >> 16);
}

// ---------------- conversion kernels ----------------
__global__ void k_cvt_bf16(const float* __restrict__ s, unsigned short* __restrict__ d, long n) {
  long i = blockIdx.x * (long)blockDim.x + threadIdx.x;
  if (i < n) d[i] = f2bf(s[i]);
}

// src [K,N] row-major f32 -> dst [N,K] row-major bf16 (weight transpose)
__global__ void k_cvtT_bf16(const float* __restrict__ s, unsigned short* __restrict__ d, int K, int N) {
  long i = blockIdx.x * (long)blockDim.x + threadIdx.x;
  if (i < (long)K * N) {
    int k = (int)(i / N), n = (int)(i % N);
    d[(long)n * K + k] = f2bf(s[i]);
  }
}

// stage softmax probabilities into bf16 with K padded to KPAD (zeros)
__global__ void k_stage_probs(const float* __restrict__ a, unsigned short* __restrict__ p) {
  long i = blockIdx.x * (long)blockDim.x + threadIdx.x;
  if (i >= (long)CB * CNH * CNP * KPAD) return;
  int mp = (int)(i % KPAD); long t = i / KPAD;
  int n = (int)(t % CNP); long bh = t / CNP;
  float v = (mp < CNP) ? a[(bh * CNP + n) * CNP + mp] : 0.f;
  p[i] = f2bf(v);
}

// stage V transposed per head: vt[b,h,d,mp] = V[b,mp,h*CC+d], K padded
__global__ void k_stage_vt(const float* __restrict__ v, unsigned short* __restrict__ vt) {
  long i = blockIdx.x * (long)blockDim.x + threadIdx.x;
  if (i >= (long)CB * CNH * CC * KPAD) return;
  int mp = (int)(i % KPAD); long t = i / KPAD;
  int d = (int)(t % CC); long bh = t / CC;
  int h = (int)(bh % CNH); int b = (int)(bh / CNH);
  float x = (mp < CNP) ? v[((long)b * CNP + mp) * CD + h * CC + d] : 0.f;
  vt[i] = f2bf(x);
}

// ------- fragment load helper: A row-major [*,ld], 16-bit A 16x32 ISA layout -------
DEV void load_fragA(FragAB& a, const unsigned short* Ap, int k0, int kh) {
  a.q[0] = *(const u32x4*)(Ap + k0 + 8 * kh);
  a.q[1] = *(const u32x4*)(Ap + k0 + 16 + 8 * kh);
}
DEV void load_fragB(FragAB& b, const unsigned short* Bp, int k0, int kh) {
  b.q[0] = *(const u32x4*)(Bp + k0 + 8 * kh);
  b.q[1] = *(const u32x4*)(Bp + k0 + 16 + 8 * kh);
}

// ---------------- bf16 WMMA GEMM: C[M,N] = A[M,K] * BT[N,K]^T + bias ----------------
__global__ __launch_bounds__(128)
void k_gemm_bf16(const unsigned short* __restrict__ A, const unsigned short* __restrict__ BT,
                 const float* __restrict__ bias, float* __restrict__ Cm,
                 int M, int N, int K, int relu)
{
  const int ntn = (N + 15) >> 4;
  const int ntm = (M + 15) >> 4;
  int tile = blockIdx.x * 4 + (threadIdx.x >> 5);     // 4 waves/block, 1 tile/wave
  if (tile >= ntm * ntn) return;                      // wave-uniform
  int tm = tile / ntn, tn = tile % ntn;
  int lane = threadIdx.x & 31;
  int kh   = lane >> 4;
  int l16  = lane & 15;
  int ar = tm * 16 + l16; if (ar >= M) ar = M - 1;    // clamp (stores guarded)
  int br = tn * 16 + l16; if (br >= N) br = N - 1;
  const unsigned short* Ap = A  + (long)ar * K;
  const unsigned short* Bp = BT + (long)br * K;
  v8f acc = {};
  for (int k0 = 0; k0 < K; k0 += 32) {
    FragAB a, b;
    load_fragA(a, Ap, k0, kh);
    load_fragB(b, Bp, k0, kh);
    if (k0 + 32 < K) __builtin_prefetch(Ap + k0 + 32, 0, 3);  // WGP-scope prefetch
    acc = __builtin_amdgcn_wmma_f32_16x16x32_bf16(false, a.bf, false, b.bf,
                                                  (short)0, acc, false, false);
  }
  int col = tn * 16 + l16;
  if (col >= N) return;
  float bs = bias ? bias[col] : 0.f;
#pragma unroll
  for (int r = 0; r < 8; ++r) {                       // D: VGPR r -> M=r+8*kh, N=l16
    int row = tm * 16 + r + 8 * kh;
    if (row < M) {
      float v = acc[r] + bs;
      if (relu) v = v > 0.f ? v : 0.f;
      Cm[(long)row * N + col] = v;
    }
  }
}

// ---- batched-per-(b,h) WMMA GEMM with independent b/h strides (attention einsums) ----
__global__ __launch_bounds__(128)
void k_gemm_bh(const unsigned short* __restrict__ A, int ldA, long sAb, long sAh,
               const unsigned short* __restrict__ BT, int ldB, long sBb, long sBh,
               float* __restrict__ Cm, int ldC, long sCb, long sCh,
               int M, int N, int K)
{
  int bh = blockIdx.y;
  int b = bh / CNH, h = bh % CNH;
  const unsigned short* Ab = A  + (long)b * sAb + (long)h * sAh;
  const unsigned short* Bb = BT + (long)b * sBb + (long)h * sBh;
  float* Cb = Cm + (long)b * sCb + (long)h * sCh;
  const int ntn = (N + 15) >> 4;
  const int ntm = (M + 15) >> 4;
  int tile = blockIdx.x * 4 + (threadIdx.x >> 5);
  if (tile >= ntm * ntn) return;
  int tm = tile / ntn, tn = tile % ntn;
  int lane = threadIdx.x & 31;
  int kh   = lane >> 4;
  int l16  = lane & 15;
  int ar = tm * 16 + l16; if (ar >= M) ar = M - 1;
  int br = tn * 16 + l16; if (br >= N) br = N - 1;
  const unsigned short* Ap = Ab + (long)ar * ldA;
  const unsigned short* Bp = Bb + (long)br * ldB;
  v8f acc = {};
  for (int k0 = 0; k0 < K; k0 += 32) {
    FragAB a, b2;
    load_fragA(a, Ap, k0, kh);
    load_fragB(b2, Bp, k0, kh);
    acc = __builtin_amdgcn_wmma_f32_16x16x32_bf16(false, a.bf, false, b2.bf,
                                                  (short)0, acc, false, false);
  }
  int col = tn * 16 + l16;
  if (col >= N) return;
#pragma unroll
  for (int r = 0; r < 8; ++r) {
    int row = tm * 16 + r + 8 * kh;
    if (row < M) Cb[(long)row * ldC + col] = acc[r];
  }
}

// ---------------- encoder helper kernels ----------------
__global__ void k_assemble_h(const float* __restrict__ proj, const float* __restrict__ bg,
                             float* __restrict__ h) {
  long i = blockIdx.x * (long)blockDim.x + threadIdx.x;
  if (i >= (long)CB * CNP * CD) return;
  int d = (int)(i % CD); long t = i / CD;
  int n = (int)(t % CNP); int b = (int)(t / CNP);
  h[i] = (n == 0) ? bg[d] : proj[((long)b * CN + (n - 1)) * CD + d];
}

// qb[b,h,n] = sum_c q[b,1+n,hc] * bp[hc]
__global__ void k_qb(const float* __restrict__ qp, const float* __restrict__ bp,
                     float* __restrict__ qb) {
  long i = blockIdx.x * (long)blockDim.x + threadIdx.x;
  if (i >= (long)CB * CNH * CN) return;
  int n = (int)(i % CN); long t = i / CN;
  int h = (int)(t % CNH); int b = (int)(t / CNH);
  const float* q = qp + ((long)(b * CNP) + n + 1) * CD + h * CC;
  float s = 0.f;
  for (int c = 0; c < CC; ++c) s += q[c] * bp[h * CC + c];
  qb[i] = s;
}

// qW[b,n,h,d] = sum_c q[b,1+n,hc] * Wp[d,hc]   (algebraic refactor of geo@Wp)
__global__ void k_qw(const float* __restrict__ qp, const float* __restrict__ Wp,
                     float* __restrict__ qw) {
  long i = blockIdx.x * (long)blockDim.x + threadIdx.x;
  if (i >= (long)CB * CN * CNH * CD) return;
  int d = (int)(i % CD); long t = i / CD;
  int h = (int)(t % CNH); t /= CNH;
  int n = (int)(t % CN);  int b = (int)(t / CN);
  const float* q = qp + ((long)(b * CNP) + n + 1) * CD + h * CC;
  const float* w = Wp + (long)d * CD + h * CC;
  float s = 0.f;
#pragma unroll
  for (int c = 0; c < CC; c += 4) {
    float4 a = *(const float4*)(q + c); float4 bb = *(const float4*)(w + c);
    s += a.x * bb.x + a.y * bb.y + a.z * bb.z + a.w * bb.w;
  }
  qw[i] = s;
}

// scores[b,h,1+n,1+m] += sum_d geo[b,n,m,d]*qW[b,n,h,d] + qb[b,h,n]
// 8 lanes cooperate on one geo row: consecutive lanes read consecutive 16B chunks
// (128B contiguous per group) -> coalesced streaming of the dominant 472MB.
__global__ __launch_bounds__(256)
void k_sp_add(const float* __restrict__ geo, const float* __restrict__ qw,
              const float* __restrict__ qb, float* __restrict__ sc) {
  long tid = blockIdx.x * (long)blockDim.x + threadIdx.x;
  long rid = tid >> 3;
  int  sub = (int)(tid & 7);
  if (rid >= (long)CB * CN * CN) return;
  int m = (int)(rid % CN); long t = rid / CN;
  int n = (int)(t % CN); int b = (int)(t / CN);
  const float* gp = geo + rid * (long)CD;
  const float* q0 = qw + (((long)b * CN + n) * CNH) * (long)CD;
  float a0 = 0, a1 = 0, a2 = 0, a3 = 0;
  for (int d = sub * 4; d < CD; d += 32) {
    float4 g4 = *(const float4*)(gp + d);
    float4 w0 = *(const float4*)(q0 + d);
    float4 w1 = *(const float4*)(q0 + CD + d);
    float4 w2 = *(const float4*)(q0 + 2 * CD + d);
    float4 w3 = *(const float4*)(q0 + 3 * CD + d);
    a0 += g4.x * w0.x + g4.y * w0.y + g4.z * w0.z + g4.w * w0.w;
    a1 += g4.x * w1.x + g4.y * w1.y + g4.z * w1.z + g4.w * w1.w;
    a2 += g4.x * w2.x + g4.y * w2.y + g4.z * w2.z + g4.w * w2.w;
    a3 += g4.x * w3.x + g4.y * w3.y + g4.z * w3.z + g4.w * w3.w;
  }
#pragma unroll
  for (int o = 1; o < 8; o <<= 1) {       // reduce within 8-lane groups (wave32)
    a0 += __shfl_xor(a0, o, 32);
    a1 += __shfl_xor(a1, o, 32);
    a2 += __shfl_xor(a2, o, 32);
    a3 += __shfl_xor(a3, o, 32);
  }
  if (sub == 0) {
    float av[4] = {a0, a1, a2, a3};
    for (int h = 0; h < CNH; ++h) {
      long o = ((long)(b * CNH + h) * CNP + (n + 1)) * CNP + (m + 1);
      sc[o] += av[h] + qb[(long)(b * CNH + h) * CN + n];
    }
  }
}

__global__ __launch_bounds__(256)
void k_softmax_rows(const float* __restrict__ in, float* __restrict__ out, int cols, float scale) {
  long row = blockIdx.x;
  const float* p = in + row * (long)cols;
  float* o = out + row * (long)cols;
  __shared__ float red[256];
  float mx = -3.4e38f;
  for (int c = threadIdx.x; c < cols; c += 256) mx = fmaxf(mx, p[c] * scale);
  red[threadIdx.x] = mx; __syncthreads();
  for (int s = 128; s > 0; s >>= 1) { if (threadIdx.x < (unsigned)s) red[threadIdx.x] = fmaxf(red[threadIdx.x], red[threadIdx.x + s]); __syncthreads(); }
  mx = red[0]; __syncthreads();
  float sum = 0.f;
  for (int c = threadIdx.x; c < cols; c += 256) sum += expf(p[c] * scale - mx);
  red[threadIdx.x] = sum; __syncthreads();
  for (int s = 128; s > 0; s >>= 1) { if (threadIdx.x < (unsigned)s) red[threadIdx.x] += red[threadIdx.x + s]; __syncthreads(); }
  sum = red[0] + 1e-20f;
  for (int c = threadIdx.x; c < cols; c += 256) o[c] = expf(p[c] * scale - mx) / sum;
}

// softmax over axis=1 of (CB, CNP, CNP): one block per (b, m)
__global__ __launch_bounds__(256)
void k_softmax_cols(const float* __restrict__ in, float* __restrict__ out) {
  int b = blockIdx.x / CNP, m = blockIdx.x % CNP;
  const float* base = in + (long)b * CNP * CNP + m;
  float* ob = out + (long)b * CNP * CNP + m;
  __shared__ float red[256];
  float mx = -3.4e38f;
  for (int n = threadIdx.x; n < CNP; n += 256) mx = fmaxf(mx, base[(long)n * CNP]);
  red[threadIdx.x] = mx; __syncthreads();
  for (int s = 128; s > 0; s >>= 1) { if (threadIdx.x < (unsigned)s) red[threadIdx.x] = fmaxf(red[threadIdx.x], red[threadIdx.x + s]); __syncthreads(); }
  mx = red[0]; __syncthreads();
  float sum = 0.f;
  for (int n = threadIdx.x; n < CNP; n += 256) sum += expf(base[(long)n * CNP] - mx);
  red[threadIdx.x] = sum; __syncthreads();
  for (int s = 128; s > 0; s >>= 1) { if (threadIdx.x < (unsigned)s) red[threadIdx.x] += red[threadIdx.x + s]; __syncthreads(); }
  sum = red[0] + 1e-20f;
  for (int n = threadIdx.x; n < CNP; n += 256) ob[(long)n * CNP] = expf(base[(long)n * CNP] - mx) / sum;
}

__global__ __launch_bounds__(256)
void k_res_ln(const float* __restrict__ a, const float* __restrict__ bv,
              const float* __restrict__ g, const float* __restrict__ be,
              float* __restrict__ out) {
  long row = blockIdx.x;
  int c = threadIdx.x;                      // CD == 256 == blockDim
  __shared__ float red[256];
  float v = a[row * CD + c] + bv[row * CD + c];
  red[c] = v; __syncthreads();
  for (int s = 128; s > 0; s >>= 1) { if (c < s) red[c] += red[c + s]; __syncthreads(); }
  float mean = red[0] / CD; __syncthreads();
  float dv = v - mean;
  red[c] = dv * dv; __syncthreads();
  for (int s = 128; s > 0; s >>= 1) { if (c < s) red[c] += red[c + s]; __syncthreads(); }
  float var = red[0] / CD; __syncthreads();
  out[row * CD + c] = dv * rsqrtf(var + 1e-5f) * g[c] + be[c];
}

__global__ __launch_bounds__(256)
void k_normalize(float* __restrict__ x) {
  long row = blockIdx.x;
  int c = threadIdx.x;
  __shared__ float red[256];
  float v = x[row * CD + c];
  red[c] = v * v; __syncthreads();
  for (int s = 128; s > 0; s >>= 1) { if (c < s) red[c] += red[c + s]; __syncthreads(); }
  float nrm = fmaxf(sqrtf(red[0]), 1e-12f);
  x[row * CD + c] = v / nrm;
}

__global__ void k_atten(const float* __restrict__ g1, const float* __restrict__ g2,
                        float* __restrict__ at) {
  long i = blockIdx.x * (long)blockDim.x + threadIdx.x;
  if (i >= (long)CB * CNP * CNP) return;
  int m = (int)(i % CNP); long t = i / CNP;
  int n = (int)(t % CNP); int b = (int)(t / CNP);
  const float* a = g1 + (long)(b * CNP + n) * CD;
  const float* c = g2 + (long)(b * CNP + m) * CD;
  float s = 0.f;
  for (int d = 0; d < CD; d += 4) {
    float4 x = *(const float4*)(a + d); float4 y = *(const float4*)(c + d);
    s += x.x * y.x + x.y * y.y + x.z * y.z + x.w * y.w;
  }
  at[i] = s / CTEMP;
}

__global__ void k_mul(const float* __restrict__ a, const float* __restrict__ b,
                      float* __restrict__ o, long n) {
  long i = blockIdx.x * (long)blockDim.x + threadIdx.x;
  if (i < n) o[i] = a[i] * b[i];
}

// ---------------- matching pipeline ----------------
__global__ void k_label(const float* __restrict__ sp, const float* __restrict__ p2,
                        int* __restrict__ fg, int* __restrict__ tidx,
                        float* __restrict__ cw, float* __restrict__ tgt) {
  long i = blockIdx.x * (long)blockDim.x + threadIdx.x;
  if (i >= (long)CB * CN) return;
  int n = (int)(i % CN), b = (int)(i / CN);
  const float* row = sp + ((long)b * CNP + n + 1) * CNP;
  int lab = 0; float best = row[0];
  for (int m = 1; m < CNP; ++m) { float v = row[m]; if (v > best) { best = v; lab = m; } }
  int f = lab > 0;
  int ti = lab - 1; if (ti < 0) ti = 0; if (ti > CN - 1) ti = CN - 1;
  fg[i] = f; tidx[i] = ti;
  cw[i] = f ? row[ti + 1] : 0.f;
  for (int d = 0; d < 3; ++d) tgt[i * 3 + d] = p2[((long)b * CN + ti) * 3 + d];
}

__global__ void k_cdf(const float* __restrict__ sp, float* __restrict__ cdf) {
  int b = blockIdx.x;
  if (threadIdx.x != 0) return;
  const float* s = sp + (long)b * CNP * CNP;
  float* c = cdf + (long)b * CN * CN;
  float acc = 0.f;
  for (int n = 0; n < CN; ++n)
    for (int m = 0; m < CN; ++m) { acc += s[(long)(n + 1) * CNP + (m + 1)]; c[(long)n * CN + m] = acc; }
  float inv = 1.f / (acc + 1e-8f);
  for (long i = 0; i < (long)CN * CN; ++i) c[i] *= inv;
}

DEV float u01(unsigned s) {
  s ^= s >> 17; s *= 0xed5ad4bbu; s ^= s >> 11; s *= 0xac4c1b51u;
  s ^= s >> 15; s *= 0x31848babu; s ^= s >> 14;
  return (float)(s >> 8) * (1.0f / 16777216.0f);
}

// Horn quaternion Kabsch
DEV void kabsch_horn(const float sc[3], const float dc[3], const float Hm[9],
                     float* __restrict__ R, float* __restrict__ t) {
  float Sxx = Hm[0], Sxy = Hm[1], Sxz = Hm[2];
  float Syx = Hm[3], Syy = Hm[4], Syz = Hm[5];
  float Szx = Hm[6], Szy = Hm[7], Szz = Hm[8];
  float K4[16] = {
    Sxx + Syy + Szz, Syz - Szy,        Szx - Sxz,         Sxy - Syx,
    Syz - Szy,       Sxx - Syy - Szz,  Sxy + Syx,         Szx + Sxz,
    Szx - Sxz,       Sxy + Syx,       -Sxx + Syy - Szz,   Syz + Szy,
    Sxy - Syx,       Szx + Sxz,        Syz + Szy,        -Sxx - Syy + Szz };
  float shift = 1e-9f;
  for (int i = 0; i < 16; ++i) shift += fabsf(K4[i]);
  for (int i = 0; i < 4; ++i) K4[i * 4 + i] += shift;
  float q[4] = {1.f, 0.017f, 0.023f, 0.031f};
  for (int it = 0; it < 40; ++it) {
    float nv[4];
    for (int r = 0; r < 4; ++r)
      nv[r] = K4[r*4+0]*q[0] + K4[r*4+1]*q[1] + K4[r*4+2]*q[2] + K4[r*4+3]*q[3];
    float nn = rsqrtf(nv[0]*nv[0] + nv[1]*nv[1] + nv[2]*nv[2] + nv[3]*nv[3] + 1e-30f);
    for (int r = 0; r < 4; ++r) q[r] = nv[r] * nn;
  }
  float w = q[0], x = q[1], y = q[2], z = q[3];
  R[0] = 1 - 2*(y*y + z*z); R[1] = 2*(x*y - w*z);     R[2] = 2*(x*z + w*y);
  R[3] = 2*(x*y + w*z);     R[4] = 1 - 2*(x*x + z*z); R[5] = 2*(y*z - w*x);
  R[6] = 2*(x*z - w*y);     R[7] = 2*(y*z + w*x);     R[8] = 1 - 2*(x*x + y*y);
  t[0] = dc[0] - (R[0]*sc[0] + R[1]*sc[1] + R[2]*sc[2]);
  t[1] = dc[1] - (R[3]*sc[0] + R[4]*sc[1] + R[5]*sc[2]);
  t[2] = dc[2] - (R[6]*sc[0] + R[7]*sc[1] + R[8]*sc[2]);
}

__global__ void k_sample(const float* __restrict__ cdf, const float* __restrict__ p1,
                         const float* __restrict__ p2, float* __restrict__ R0,
                         float* __restrict__ t0) {
  long i = blockIdx.x * (long)blockDim.x + threadIdx.x;
  if (i >= (long)CB * NP1) return;
  int b = (int)(i / NP1), k = (int)(i % NP1);
  const float* cd = cdf + (long)b * CN * CN;
  float sp1[3][3], sp2[3][3];
  for (int j = 0; j < 3; ++j) {
    float u = u01((unsigned)((b * 3 * NP1 + k * 3 + j) * 2654435761u + 12345u));
    int lo = 0, hi = CN * CN;                       // searchsorted 'left'
    while (lo < hi) { int mid = (lo + hi) >> 1; if (cd[mid] < u) lo = mid + 1; else hi = mid; }
    int idx = lo; if (idx > CN * CN - 1) idx = CN * CN - 1;
    int i1 = idx / CN, i2 = idx % CN;
    for (int d = 0; d < 3; ++d) {
      sp1[j][d] = p1[((long)b * CN + i1) * 3 + d];
      sp2[j][d] = p2[((long)b * CN + i2) * 3 + d];
    }
  }
  float sc[3] = {0, 0, 0}, dc[3] = {0, 0, 0};
  for (int j = 0; j < 3; ++j)
    for (int d = 0; d < 3; ++d) { sc[d] += sp2[j][d] * (1.f/3.f); dc[d] += sp1[j][d] * (1.f/3.f); }
  float Hm[9] = {0};
  for (int j = 0; j < 3; ++j)
    for (int a = 0; a < 3; ++a)
      for (int c = 0; c < 3; ++c)
        Hm[a * 3 + c] += (sp2[j][a] - sc[a]) * (sp1[j][c] - dc[c]) * (1.f/3.f);
  kabsch_horn(sc, dc, Hm, R0 + i * 9, t0 + i * 3);
}

__global__ void k_hyp(const float* __restrict__ R0, const float* __restrict__ t0,
                      const float* __restrict__ tgt, const float* __restrict__ cw,
                      const float* __restrict__ p1, float* __restrict__ hyp) {
  long i = blockIdx.x * (long)blockDim.x + threadIdx.x;
  if (i >= (long)CB * NP1) return;
  int b = (int)(i / NP1);
  const float* R = R0 + i * 9; const float* t = t0 + i * 3;
  float s = 0.f;
  for (int n = 0; n < CN; ++n) {
    const float* g = tgt + ((long)b * CN + n) * 3;
    const float* p = p1 + ((long)b * CN + n) * 3;
    float x = R[0]*g[0] + R[1]*g[1] + R[2]*g[2] + t[0] - p[0];
    float y = R[3]*g[0] + R[4]*g[1] + R[5]*g[2] + t[1] - p[1];
    float z = R[6]*g[0] + R[7]*g[1] + R[8]*g[2] + t[2] - p[2];
    if (x*x + y*y + z*z < CTAU * CTAU) s += cw[(long)b * CN + n];
  }
  hyp[i] = s;
}

__global__ void k_rank(const float* __restrict__ hyp, int* __restrict__ top) {
  long i = blockIdx.x * (long)blockDim.x + threadIdx.x;
  if (i >= (long)CB * NP1) return;
  int b = (int)(i / NP1), k = (int)(i % NP1);
  const float* hb = hyp + (long)b * NP1;
  float v = hb[k]; int r = 0;
  for (int kk = 0; kk < NP1; ++kk) { float o = hb[kk]; r += (o > v) || (o == v && kk < k); }
  if (r < NP2) top[b * NP2 + r] = k;   // descending, stable ties
}

__global__ void k_refine(const int* __restrict__ top, const float* __restrict__ R0,
                         const float* __restrict__ t0, const float* __restrict__ tgt,
                         const float* __restrict__ cw, const float* __restrict__ p1,
                         float* __restrict__ Rr, float* __restrict__ tr2) {
  long i = blockIdx.x * (long)blockDim.x + threadIdx.x;
  if (i >= (long)CB * NP2) return;
  int b = (int)(i / NP2);
  int k = top[i];
  const float* R = R0 + ((long)b * NP1 + k) * 9;
  const float* t = t0 + ((long)b * NP1 + k) * 3;
  float Sw = 0.f, ss[3] = {0,0,0}, dd[3] = {0,0,0};
  for (int n = 0; n < CN; ++n) {
    const float* g = tgt + ((long)b * CN + n) * 3;
    const float* p = p1 + ((long)b * CN + n) * 3;
    float x = R[0]*g[0] + R[1]*g[1] + R[2]*g[2] + t[0] - p[0];
    float y = R[3]*g[0] + R[4]*g[1] + R[5]*g[2] + t[1] - p[1];
    float z = R[6]*g[0] + R[7]*g[1] + R[8]*g[2] + t[2] - p[2];
    float w = ((x*x + y*y + z*z < CTAU * CTAU) ? cw[(long)b * CN + n] : 0.f) + 1e-8f;
    Sw += w;
    for (int d = 0; d < 3; ++d) { ss[d] += w * g[d]; dd[d] += w * p[d]; }
  }
  float inv = 1.f / (Sw + 1e-8f);
  float sc[3] = {ss[0]*inv, ss[1]*inv, ss[2]*inv};
  float dc[3] = {dd[0]*inv, dd[1]*inv, dd[2]*inv};
  float Hm[9] = {0};
  for (int n = 0; n < CN; ++n) {
    const float* g = tgt + ((long)b * CN + n) * 3;
    const float* p = p1 + ((long)b * CN + n) * 3;
    float x = R[0]*g[0] + R[1]*g[1] + R[2]*g[2] + t[0] - p[0];
    float y = R[3]*g[0] + R[4]*g[1] + R[5]*g[2] + t[1] - p[1];
    float z = R[6]*g[0] + R[7]*g[1] + R[8]*g[2] + t[2] - p[2];
    float w = (((x*x + y*y + z*z < CTAU * CTAU) ? cw[(long)b * CN + n] : 0.f) + 1e-8f) * inv;
    for (int a = 0; a < 3; ++a)
      for (int c = 0; c < 3; ++c)
        Hm[a * 3 + c] += w * (g[a] - sc[a]) * (p[c] - dc[c]);
  }
  kabsch_horn(sc, dc, Hm, Rr + i * 9, tr2 + i * 3);
}

__global__ __launch_bounds__(256)
void k_chamfer(const float* __restrict__ Rr, const float* __restrict__ tr2,
               const float* __restrict__ model, const float* __restrict__ radius,
               const float* __restrict__ p1, const int* __restrict__ fg,
               float* __restrict__ cost) {
  int bj = blockIdx.x;
  int b = bj / NP2;
  const float* R = Rr + (long)bj * 9; const float* t = tr2 + (long)bj * 3;
  float inv = 1.f / (radius[b] + 1e-6f);
  __shared__ float red[256];
  float acc = 0.f;
  for (int m = threadIdx.x; m < CNM; m += 256) {
    const float* mp = model + ((long)b * CNM + m) * 3;
    float mx = mp[0] * inv, my = mp[1] * inv, mz = mp[2] * inv;
    float tx = R[0]*mx + R[1]*my + R[2]*mz + t[0];
    float ty = R[3]*mx + R[4]*my + R[5]*mz + t[1];
    float tz = R[6]*mx + R[7]*my + R[8]*mz + t[2];
    float mind = 3.4e38f;
    for (int n = 0; n < CN; ++n) {
      if (!fg[(long)b * CN + n]) continue;
      const float* p = p1 + ((long)b * CN + n) * 3;
      float dx = tx - p[0], dy = ty - p[1], dz = tz - p[2];
      mind = fminf(mind, dx*dx + dy*dy + dz*dz);
    }
    acc += sqrtf(mind + 1e-12f);
  }
  red[threadIdx.x] = acc; __syncthreads();
  for (int s = 128; s > 0; s >>= 1) { if (threadIdx.x < (unsigned)s) red[threadIdx.x] += red[threadIdx.x + s]; __syncthreads(); }
  if (threadIdx.x == 0) cost[bj] = red[0] / (float)CNM;
}

__global__ void k_final(const float* __restrict__ cost, const float* __restrict__ Rr,
                        const float* __restrict__ tr2, float* __restrict__ out) {
  int b = threadIdx.x;
  if (b >= CB) return;
  int best = 0; float bc = cost[(long)b * NP2];
  for (int j = 1; j < NP2; ++j) { float c = cost[(long)b * NP2 + j]; if (c < bc) { bc = c; best = j; } }
  for (int i = 0; i < 9; ++i) out[b * 9 + i] = Rr[((long)b * NP2 + best) * 9 + i];
  for (int i = 0; i < 3; ++i) out[CB * 9 + b * 3 + i] = tr2[((long)b * NP2 + best) * 3 + i];
}

// ---------------- input index map ----------------
enum {
  IN_P1 = 0, IN_F1, IN_GEO1, IN_P2, IN_F2, IN_GEO2, IN_RADIUS, IN_MODEL,
  P_IN_W, P_IN_B, P_OUT_W, P_OUT_B, P_BG,
  PS_WQ, PS_BQ, PS_WK, PS_BK, PS_WV, PS_BV, PS_WP, PS_BP, PS_WO, PS_BO,
  PS_W1, PS_B1, PS_W2, PS_B2, PS_G1, PS_G2, PS_BE1, PS_BE2,
  PC_WQ, PC_BQ, PC_WK, PC_BK, PC_WV, PC_BV, PC_WO, PC_BO,
  PC_W1, PC_B1, PC_W2, PC_B2, PC_G1, PC_G2, PC_BE1, PC_BE2
};

extern "C" void kernel_launch(void* const* d_in, const int* in_sizes, int n_in,
                              void* d_out, int out_size, void* d_ws, size_t ws_size,
                              hipStream_t stream) {
  (void)in_sizes; (void)n_in; (void)out_size; (void)ws_size;
  auto F = [&](int i) { return (const float*)d_in[i]; };

  char* wsp = (char*)d_ws;
  size_t off = 0;
  auto alloc = [&](size_t bytes) -> void* {
    void* p = wsp + off;
    off = (off + bytes + 255) & ~(size_t)255;
    return p;
  };
  auto allocF  = [&](long n) { return (float*)alloc(sizeof(float) * n); };
  auto allocU  = [&](long n) { return (unsigned short*)alloc(sizeof(unsigned short) * n); };
  auto allocI  = [&](long n) { return (int*)alloc(sizeof(int) * n); };

  // bf16 weights (transposed to [N,K])
  unsigned short* inWT  = allocU((long)CD * CD);
  unsigned short* outWT = allocU((long)CD * CD);
  unsigned short *WqT[2][3], *WkT[2][3], *WvT[2][3], *WoT[2][3], *W1T[2][3], *W2T[2][3];
  for (int kd = 0; kd < 2; ++kd)
    for (int bk = 0; bk < 3; ++bk) {
      WqT[kd][bk] = allocU((long)CD * CD); WkT[kd][bk] = allocU((long)CD * CD);
      WvT[kd][bk] = allocU((long)CD * CD); WoT[kd][bk] = allocU((long)CD * CD);
      W1T[kd][bk] = allocU((long)CD * CF); W2T[kd][bk] = allocU((long)CF * CD);
    }
  // activation buffers
  unsigned short* abf  = allocU((long)CB * CNP * CF);           // bf16 GEMM A staging
  unsigned short* qbf  = allocU((long)CB * CNP * CD);           // bf16 Q for scores WMMA
  unsigned short* kbf  = allocU((long)CB * CNP * CD);           // bf16 K for scores WMMA
  unsigned short* pbf  = allocU((long)CB * CNH * CNP * KPAD);   // bf16 probs, K-padded
  unsigned short* vtbf = allocU((long)CB * CNH * CC * KPAD);    // bf16 V^T, K-padded
  float* h1 = allocF((long)CB * CNP * CD);
  float* h2 = allocF((long)CB * CNP * CD);
  float* projb = allocF((long)CB * CN * CD);
  float* qp = allocF((long)CB * CNP * CD);
  float* kp = allocF((long)CB * CNP * CD);
  float* vp = allocF((long)CB * CNP * CD);
  float* obuf  = allocF((long)CB * CNP * CD);
  float* oproj = allocF((long)CB * CNP * CD);
  float* tmpx  = allocF((long)CB * CNP * CD);
  float* ff1   = allocF((long)CB * CNP * CF);
  float* ff2   = allocF((long)CB * CNP * CD);
  float* scores = allocF((long)CB * CNH * CNP * CNP);
  float* qwbuf  = allocF((long)CB * CN * CNH * CD);
  float* qbbuf  = allocF((long)CB * CNH * CN);
  float* g1n = allocF((long)CB * CNP * CD);
  float* g2n = allocF((long)CB * CNP * CD);
  float* atten  = allocF((long)CB * CNP * CNP);
  float* srow   = allocF((long)CB * CNP * CNP);
  float* scol   = allocF((long)CB * CNP * CNP);
  float* scoreP = allocF((long)CB * CNP * CNP);
  int*   fgI   = allocI((long)CB * CN);
  int*   tidxI = allocI((long)CB * CN);
  float* cw    = allocF((long)CB * CN);
  float* tgt   = allocF((long)CB * CN * 3);
  float* cdf   = allocF((long)CB * CN * CN);
  float* R0 = allocF((long)CB * NP1 * 9);
  float* t0 = allocF((long)CB * NP1 * 3);
  float* hyp = allocF((long)CB * NP1);
  int*   topI = allocI((long)CB * NP2);
  float* Rr  = allocF((long)CB * NP2 * 9);
  float* tr2 = allocF((long)CB * NP2 * 3);
  float* cost = allocF((long)CB * NP2);

#define GRID1(n) dim3((unsigned)(((n) + 255) / 256)), dim3(256), 0, stream

  // ---- convert weights ----
  auto cvtT = [&](const float* src, unsigned short* dst, int K, int N) {
    long n = (long)K * N;
    hipLaunchKernelGGL(k_cvtT_bf16, GRID1(n), src, dst, K, N);
  };
  cvtT(F(P_IN_W), inWT, CD, CD);
  cvtT(F(P_OUT_W), outWT, CD, CD);
  const int iWq[2] = {PS_WQ, PC_WQ}, iWk[2] = {PS_WK, PC_WK}, iWv[2] = {PS_WV, PC_WV};
  const int iWo[2] = {PS_WO, PC_WO}, iW1[2] = {PS_W1, PC_W1}, iW2[2] = {PS_W2, PC_W2};
  const int iBq[2] = {PS_BQ, PC_BQ}, iBk[2] = {PS_BK, PC_BK}, iBv[2] = {PS_BV, PC_BV};
  const int iBo[2] = {PS_BO, PC_BO}, iB1[2] = {PS_B1, PC_B1}, iB2[2] = {PS_B2, PC_B2};
  const int iG1[2] = {PS_G1, PC_G1}, iG2[2] = {PS_G2, PC_G2};
  const int iE1[2] = {PS_BE1, PC_BE1}, iE2[2] = {PS_BE2, PC_BE2};
  for (int kd = 0; kd < 2; ++kd)
    for (int bk = 0; bk < 3; ++bk) {
      cvtT(F(iWq[kd]) + (long)bk * CD * CD, WqT[kd][bk], CD, CD);
      cvtT(F(iWk[kd]) + (long)bk * CD * CD, WkT[kd][bk], CD, CD);
      cvtT(F(iWv[kd]) + (long)bk * CD * CD, WvT[kd][bk], CD, CD);
      cvtT(F(iWo[kd]) + (long)bk * CD * CD, WoT[kd][bk], CD, CD);
      cvtT(F(iW1[kd]) + (long)bk * CD * CF, W1T[kd][bk], CD, CF);
      cvtT(F(iW2[kd]) + (long)bk * CF * CD, W2T[kd][bk], CF, CD);
    }

  // ---- GEMM wrapper ----
  auto gemm = [&](const float* Af, long M, int K, const unsigned short* BT,
                  const float* bias, float* Cout, int N, int relu) {
    long n = M * (long)K;
    hipLaunchKernelGGL(k_cvt_bf16, GRID1(n), Af, abf, n);
    long tiles = ((M + 15) / 16) * ((N + 15) / 16);
    hipLaunchKernelGGL(k_gemm_bf16, dim3((unsigned)((tiles + 3) / 4)), dim3(128), 0, stream,
                       abf, BT, bias, Cout, (int)M, N, K, relu);
  };

  // ---- one attention block ----
  auto attn = [&](float* xq, const float* xkv, const float* geo, int kd, int bk) {
    gemm(xq,  (long)CB * CNP, CD, WqT[kd][bk], F(iBq[kd]) + bk * CD, qp, CD, 0);
    gemm(xkv, (long)CB * CNP, CD, WkT[kd][bk], F(iBk[kd]) + bk * CD, kp, CD, 0);
    gemm(xkv, (long)CB * CNP, CD, WvT[kd][bk], F(iBv[kd]) + bk * CD, vp, CD, 0);
    // scores = Q K^T via batched WMMA (K's row-major storage IS B^T; contraction = head dim)
    hipLaunchKernelGGL(k_cvt_bf16, GRID1((long)CB * CNP * CD), qp, qbf, (long)CB * CNP * CD);
    hipLaunchKernelGGL(k_cvt_bf16, GRID1((long)CB * CNP * CD), kp, kbf, (long)CB * CNP * CD);
    {
      int ntm = (CNP + 15) / 16, ntn = (CNP + 15) / 16;
      dim3 grid((unsigned)((ntm * ntn + 3) / 4), CB * CNH);
      hipLaunchKernelGGL(k_gemm_bh, grid, dim3(128), 0, stream,
                         qbf, CD, (long)CNP * CD, (long)CC,
                         kbf, CD, (long)CNP * CD, (long)CC,
                         scores, CNP, (long)CNH * CNP * CNP, (long)CNP * CNP,
                         CNP, CNP, CC);
    }
    if (geo) {
      const float* Wp = F(PS_WP) + (long)bk * CD * CD;
      const float* bp = F(PS_BP) + bk * CD;
      hipLaunchKernelGGL(k_qb, GRID1((long)CB * CNH * CN), qp, bp, qbbuf);
      hipLaunchKernelGGL(k_qw, GRID1((long)CB * CN * CNH * CD), qp, Wp, qwbuf);
      hipLaunchKernelGGL(k_sp_add, GRID1((long)CB * CN * CN * 8), geo, qwbuf, qbbuf, scores);
    }
    hipLaunchKernelGGL(k_softmax_rows, dim3(CB * CNH * CNP), dim3(256), 0, stream,
                       scores, scores, CNP, 0.125f);   // 1/sqrt(C)
    // O = A V via batched WMMA: stage probs (K-padded) and V^T
    hipLaunchKernelGGL(k_stage_probs, GRID1((long)CB * CNH * CNP * KPAD), scores, pbf);
    hipLaunchKernelGGL(k_stage_vt, GRID1((long)CB * CNH * CC * KPAD), vp, vtbf);
    {
      int ntm = (CNP + 15) / 16, ntn = (CC + 15) / 16;
      dim3 grid((unsigned)((ntm * ntn + 3) / 4), CB * CNH);
      hipLaunchKernelGGL(k_gemm_bh, grid, dim3(128), 0, stream,
                         pbf, KPAD, (long)CNH * CNP * KPAD, (long)CNP * KPAD,
                         vtbf, KPAD, (long)CNH * CC * KPAD, (long)CC * KPAD,
                         obuf, CD, (long)CNP * CD, (long)CC,
                         CNP, CC, KPAD);
    }
    gemm(obuf, (long)CB * CNP, CD, WoT[kd][bk], F(iBo[kd]) + bk * CD, oproj, CD, 0);
    hipLaunchKernelGGL(k_res_ln, dim3(CB * CNP), dim3(256), 0, stream,
                       xq, oproj, F(iG1[kd]) + bk * CD, F(iE1[kd]) + bk * CD, tmpx);
    gemm(tmpx, (long)CB * CNP, CD, W1T[kd][bk], F(iB1[kd]) + bk * CF, ff1, CF, 1);
    gemm(ff1,  (long)CB * CNP, CF, W2T[kd][bk], F(iB2[kd]) + bk * CD, ff2, CD, 0);
    hipLaunchKernelGGL(k_res_ln, dim3(CB * CNP), dim3(256), 0, stream,
                       tmpx, ff2, F(iG2[kd]) + bk * CD, F(iE2[kd]) + bk * CD, xq);
  };

  // ---- input projection + bg token ----
  gemm(F(IN_F1), (long)CB * CN, CD, inWT, F(P_IN_B), projb, CD, 0);
  hipLaunchKernelGGL(k_assemble_h, GRID1((long)CB * CNP * CD), projb, F(P_BG), h1);
  gemm(F(IN_F2), (long)CB * CN, CD, inWT, F(P_IN_B), projb, CD, 0);
  hipLaunchKernelGGL(k_assemble_h, GRID1((long)CB * CNP * CD), projb, F(P_BG), h2);

  // ---- encoder ----
  for (int bk = 0; bk < 3; ++bk) {
    attn(h1, h1, F(IN_GEO1), 0, bk);
    attn(h2, h2, F(IN_GEO2), 0, bk);
    attn(h1, h2, nullptr, 1, bk);
    attn(h2, h1, nullptr, 1, bk);
  }

  // ---- output features + similarity ----
  gemm(h1, (long)CB * CNP, CD, outWT, F(P_OUT_B), g1n, CD, 0);
  hipLaunchKernelGGL(k_normalize, dim3(CB * CNP), dim3(256), 0, stream, g1n);
  gemm(h2, (long)CB * CNP, CD, outWT, F(P_OUT_B), g2n, CD, 0);
  hipLaunchKernelGGL(k_normalize, dim3(CB * CNP), dim3(256), 0, stream, g2n);
  hipLaunchKernelGGL(k_atten, GRID1((long)CB * CNP * CNP), g1n, g2n, atten);

  // ---- coarse R,t ----
  hipLaunchKernelGGL(k_softmax_rows, dim3(CB * CNP), dim3(256), 0, stream, atten, srow, CNP, 1.0f);
  hipLaunchKernelGGL(k_softmax_cols, dim3(CB * CNP), dim3(256), 0, stream, atten, scol);
  hipLaunchKernelGGL(k_mul, GRID1((long)CB * CNP * CNP), srow, scol, scoreP, (long)CB * CNP * CNP);
  hipLaunchKernelGGL(k_label, GRID1((long)CB * CN), scoreP, F(IN_P2), fgI, tidxI, cw, tgt);
  hipLaunchKernelGGL(k_cdf, dim3(CB), dim3(32), 0, stream, scoreP, cdf);
  hipLaunchKernelGGL(k_sample, GRID1((long)CB * NP1), cdf, F(IN_P1), F(IN_P2), R0, t0);
  hipLaunchKernelGGL(k_hyp, GRID1((long)CB * NP1), R0, t0, tgt, cw, F(IN_P1), hyp);
  hipLaunchKernelGGL(k_rank, GRID1((long)CB * NP1), hyp, topI);
  hipLaunchKernelGGL(k_refine, GRID1((long)CB * NP2), topI, R0, t0, tgt, cw, F(IN_P1), Rr, tr2);
  hipLaunchKernelGGL(k_chamfer, dim3(CB * NP2), dim3(256), 0, stream,
                     Rr, tr2, F(IN_MODEL), F(IN_RADIUS), F(IN_P1), fgI, cost);
  hipLaunchKernelGGL(k_final, dim3(1), dim3(32), 0, stream, cost, Rr, tr2, (float*)d_out);
#undef GRID1
}
